// MPNN_55705725829535
// MI455X (gfx1250) — matrix operand order repaired and checked by reading, not compiled, not run on hardware
//
#include <hip/hip_runtime.h>
#include <math.h>

#define N_NODES 10000
#define N_EDGES 50000
#define N_GRAPHS 64
#define NODE_DIM 32
#define EDGE_DIM 16
#define HID 64
#define EDGE_HID 128
#define T_STEPS 3
#define S2S_STEPS 6
#define N_ETILES (N_EDGES / 16)   // 3125

typedef __attribute__((ext_vector_type(16))) __bf16 v16bf;
typedef __attribute__((ext_vector_type(8)))  float  v8f;
typedef unsigned int u32x4 __attribute__((ext_vector_type(4)));
typedef int          i32x4 __attribute__((ext_vector_type(4)));
typedef int          i32x8 __attribute__((ext_vector_type(8)));

#define Z8 {0.f,0.f,0.f,0.f,0.f,0.f,0.f,0.f}

__device__ inline unsigned short f32_to_bf16(float f) {
  unsigned int u = __float_as_uint(f);
  unsigned int lsb = (u >> 16) & 1u;
  u += 0x7fffu + lsb;                 // round-to-nearest-even
  return (unsigned short)(u >> 16);
}
__device__ inline float bf16_to_f32(unsigned short s) {
  return __uint_as_float(((unsigned int)s) << 16);
}
__device__ inline float sigmoidf_(float x) { return 1.f / (1.f + __expf(-x)); }

__device__ inline v8f wmma_bf16(v16bf a, v16bf b, v8f c) {
  return __builtin_amdgcn_wmma_f32_16x16x32_bf16(false, a, false, b, (short)0, c, false, false);
}

// ---- Tensor Data Mover: 1-D bulk copy global -> LDS (data_size = 8B units) -
// D# per CDNA5 ISA ch.8: group0 = {count=1 | lds_addr | global_addr | type=2},
// group1 = {data_size=3(8B); tensor_dim0=tile_dim0=units; tensor_dim1=1}.
__device__ inline void tdm_load_1d(unsigned lds_addr, const void* gsrc, unsigned units8) {
  unsigned long long ga = (unsigned long long)(size_t)gsrc;
  u32x4 g0 = { 1u,                                   // count=1, user mode
               lds_addr,                              // LDS byte address
               (unsigned)ga,                          // global_addr[31:0]
               ((unsigned)(ga >> 32) & 0x01ffffffu) | 0x80000000u }; // [56:32] | type=2
  i32x8 g1 = { (int)(3u << 16),                      // data_size = 8B, no multicast
               (int)(units8 << 16),                  // tensor_dim0[15:0]  (bits 79:48)
               (int)(1u << 16),                      // tensor_dim1 = 1    (bits 111:80)
               (int)(units8 << 16),                  // tile_dim0          (bits 127:112)
               0,                                    // tile_dim1/2 = 0 (1-D)
               (int)units8,                          // tensor_dim0_stride low
               0, 0 };
  i32x4 gz = { 0, 0, 0, 0 };
#if defined(__clang_major__) && (__clang_major__ >= 23)
  i32x8 gz8 = { 0, 0, 0, 0, 0, 0, 0, 0 };
  __builtin_amdgcn_tensor_load_to_lds(g0, g1, gz, gz, gz8, 0);
#else
  __builtin_amdgcn_tensor_load_to_lds(g0, g1, gz, gz, 0);
#endif
}

// ---- CDNA5 wave32 WMMA operand packing -------------------------------------
// A (16x32 bf16): lane<16 -> row=lane, halves 0..7 = K 0..7, halves 8..15 = K 16..23
//                 lane>=16 -> row=lane-16, K offsets +8 (K 8..15 / 24..31)
__device__ inline v16bf make_apack_f32(const float* __restrict__ src, int rowStride,
                                       int kc, int lane, int maxK) {
  union { v16bf v; unsigned short u[16]; } r;
  int m  = lane & 15;
  int hi = (lane >> 4) & 1;
  const float* row = src + (size_t)m * rowStride;
#pragma unroll
  for (int h = 0; h < 16; ++h) {
    int k = kc * 32 + ((h < 8) ? h : h + 8) + (hi ? 8 : 0);
    float v = (k < maxK) ? row[k] : 0.f;
    r.u[h] = f32_to_bf16(v);
  }
  return r.v;
}
__device__ inline v16bf make_apack_bf16(const unsigned short* src,
                                        int rowStride, int kc, int lane) {
  union { v16bf v; unsigned short u[16]; } r;
  int m  = lane & 15;
  int hi = (lane >> 4) & 1;
  const unsigned short* row = src + (size_t)m * rowStride;
#pragma unroll
  for (int h = 0; h < 16; ++h) {
    int k = kc * 32 + ((h < 8) ? h : h + 8) + (hi ? 8 : 0);
    r.u[h] = row[k];
  }
  return r.v;
}
// B (32x16 bf16): lane = N + (((K/8)&1)?16:0); half h = (K&7) + ((K>=16)?8:0). Pre-packed so each
// lane's 16 halves are 32 contiguous bytes.
__device__ inline v16bf load_bpack(const unsigned short* __restrict__ pk,
                                   int nt, int kc, int KCH, int lane) {
  const v16bf* q = (const v16bf*)(pk + (((size_t)(nt * KCH + kc)) * 32 + lane) * 16);
  return *q;
}

// Pre-pack a row-major f32 weight (R x C) into WMMA-B bf16 layout.
// trans=0: B[k][n]=W[k*C+n] (K=R->padK, N=C). trans=1: B=W^T (K=C->padK, N=R).
__global__ void pack_b_kernel(const float* __restrict__ W, unsigned short* __restrict__ out,
                              int R, int C, int trans, int padK) {
  int idx = blockIdx.x * blockDim.x + threadIdx.x;
  int N   = trans ? R : C;
  int NT  = N / 16, KCH = padK / 32;
  int total = NT * KCH * 512;
  if (idx >= total) return;
  int half = idx & 15;
  int lane = (idx >> 4) & 31;
  int rest = idx >> 9;
  int kc   = rest % KCH;
  int nt   = rest / KCH;
  int k = kc * 32 + ((lane >= 16) ? 8 : 0) + ((half < 8) ? half : half + 8);
  int n = nt * 16 + (lane & 15);
  int realK = trans ? C : R;
  float v = 0.f;
  if (k < realK) v = trans ? W[(size_t)n * C + k] : W[(size_t)k * C + n];
  out[idx] = f32_to_bf16(v);
}

__global__ void zero_kernel(float* p, int n) {
  int i = blockIdx.x * blockDim.x + threadIdx.x;
  if (i < n) p[i] = 0.f;
}

// ---- h0 = node_features @ W_proj + b_proj  (WMMA, K=32 exact) --------------
__global__ void node_proj_kernel(const float* __restrict__ nf, const float* __restrict__ bproj,
                                 const unsigned short* __restrict__ Wpk, float* __restrict__ h) {
  int tile = blockIdx.x, lane = threadIdx.x;
  const float* src = nf + (size_t)tile * 16 * NODE_DIM;
  v16bf a = make_apack_f32(src, NODE_DIM, 0, lane, NODE_DIM);
  int rbase = (lane >= 16) ? 8 : 0;
#pragma unroll
  for (int nt = 0; nt < 4; ++nt) {
    v8f c = Z8;
    c = wmma_bf16(a, load_bpack(Wpk, nt, 0, 1, lane), c);
    int col = nt * 16 + (lane & 15);
    float bias = bproj[col];
#pragma unroll
    for (int v = 0; v < 8; ++v) {
      int node = tile * 16 + rbase + v;
      h[(size_t)node * HID + col] = c[v] + bias;
    }
  }
}

// ---- hidden = relu(edge_features @ W_e1 + b_e1) -> bf16 (K padded 16->32) --
__global__ void edge_hidden_kernel(const float* __restrict__ ef, const float* __restrict__ be1,
                                   const unsigned short* __restrict__ Wpk,
                                   unsigned short* __restrict__ hidden) {
  int tile = blockIdx.x, lane = threadIdx.x;
  const float* src = ef + (size_t)tile * 16 * EDGE_DIM;
  v16bf a = make_apack_f32(src, EDGE_DIM, 0, lane, EDGE_DIM);
  int rbase = (lane >= 16) ? 8 : 0;
#pragma unroll
  for (int nt = 0; nt < 8; ++nt) {
    v8f c = Z8;
    c = wmma_bf16(a, load_bpack(Wpk, nt, 0, 1, lane), c);
    int col = nt * 16 + (lane & 15);
    float bias = be1[col];
#pragma unroll
    for (int v = 0; v < 8; ++v) {
      float x = c[v] + bias;
      x = x > 0.f ? x : 0.f;
      hidden[(size_t)(tile * 16 + rbase + v) * EDGE_HID + col] = f32_to_bf16(x);
    }
  }
}

// ---- A = hidden @ W_e2 + b_e2 -> bf16 (E x 4096). The 52-GFLOP GEMM. -------
// Block = 8 waves = 8 edge tiles sharing ONE 64-col slice of W_e2.
// TDM (tensor_load_to_lds) DMAs the 16KB B slice + 32KB of `hidden` into LDS
// once per block; waves run WMMA from LDS, stage f32->bf16 tiles in LDS, and
// write A with coalesced b128 stores.
__global__ void edge_A_kernel(const unsigned short* __restrict__ hidden,
                              const unsigned short* __restrict__ We2pk,
                              const float* __restrict__ be2,
                              unsigned short* __restrict__ A) {
  __shared__ __align__(32) unsigned short bbuf[4 * 2048];   // 16KB: 4 nt x 4 kc x 512
  __shared__ __align__(32) unsigned short hbuf[8 * 2048];   // 32KB: 8 edge tiles x 16x128
  __shared__ __align__(16) unsigned short stage[8][16 * 64]; // 16KB
  int wid  = threadIdx.x >> 5;
  int lane = threadIdx.x & 31;
  int s   = blockIdx.y;                 // 64 col-slices of 64 cols
  int et0 = blockIdx.x * 8;
  if (wid == 0) {
    // one DMA pair per block (TENSORcnt -> 2), issued by wave 0 only
    tdm_load_1d((unsigned)(size_t)(void*)bbuf, We2pk + (size_t)s * 8192, 2048u);
    int remTiles = N_ETILES - et0; if (remTiles > 8) remTiles = 8;
    tdm_load_1d((unsigned)(size_t)(void*)hbuf, hidden + (size_t)et0 * 2048,
                (unsigned)remTiles * 512u);
    __builtin_amdgcn_s_wait_tensorcnt(0);
  }
  __syncthreads();
  int et = et0 + wid;
  if (et >= N_ETILES) return;
  const unsigned short* hsrc = hbuf + wid * 2048;  // LDS, 16 rows x 128 bf16
  v16bf a0 = make_apack_bf16(hsrc, EDGE_HID, 0, lane);
  v16bf a1 = make_apack_bf16(hsrc, EDGE_HID, 1, lane);
  v16bf a2 = make_apack_bf16(hsrc, EDGE_HID, 2, lane);
  v16bf a3 = make_apack_bf16(hsrc, EDGE_HID, 3, lane);
  unsigned short* st = stage[wid];
  int rbase = (lane >= 16) ? 8 : 0;
#pragma unroll
  for (int q = 0; q < 4; ++q) {
    int nt = s * 4 + q;
    v8f c = Z8;
    c = wmma_bf16(a0, *(const v16bf*)(bbuf + ((q * 4 + 0) * 32 + lane) * 16), c);
    c = wmma_bf16(a1, *(const v16bf*)(bbuf + ((q * 4 + 1) * 32 + lane) * 16), c);
    c = wmma_bf16(a2, *(const v16bf*)(bbuf + ((q * 4 + 2) * 32 + lane) * 16), c);
    c = wmma_bf16(a3, *(const v16bf*)(bbuf + ((q * 4 + 3) * 32 + lane) * 16), c);
    int col = lane & 15;
    float bias = be2[nt * 16 + col];
#pragma unroll
    for (int v = 0; v < 8; ++v)
      st[(rbase + v) * 64 + q * 16 + col] = f32_to_bf16(c[v] + bias);
  }
  asm volatile("s_wait_dscnt 0x0" ::: "memory");  // wave-local LDS RAW fence
  const uint4* src4 = (const uint4*)st;           // 16 rows x 128B = 128 chunks
#pragma unroll
  for (int p2 = 0; p2 < 4; ++p2) {
    int o   = lane + p2 * 32;
    int row = o >> 3;
    int cs  = (o & 7) * 8;
    uint4 d = src4[o];
    *(uint4*)(A + (size_t)(et * 16 + row) * 4096 + s * 64 + cs) = d;
  }
}

// ---- msgs = einsum('eij,ej->ei'), scatter-add to m[dst] (HBM-bound on A) ---
// One wave per edge; lane computes output rows i=lane and i=lane+32.
__global__ void message_kernel(const unsigned short* __restrict__ A,
                               const float* __restrict__ h,
                               const int* __restrict__ edge_index,
                               float* __restrict__ m) {
  int wid  = threadIdx.x >> 5;
  int lane = threadIdx.x & 31;
  int e = blockIdx.x * 8 + wid;
  int dst = edge_index[e];
  int src = edge_index[N_EDGES + e];
  const unsigned short* Ae = A + (size_t)e * 4096;
  const float* hs = h + (size_t)src * HID;
  float acc0 = 0.f, acc1 = 0.f;
#pragma unroll 4
  for (int j = 0; j < HID; j += 4) {
    float4 hv = *(const float4*)(hs + j);               // broadcast across lanes
    uint2 u0 = *(const uint2*)(Ae + (size_t)lane * 64 + j);
    uint2 u1 = *(const uint2*)(Ae + (size_t)(lane + 32) * 64 + j);
    acc0 += bf16_to_f32((unsigned short)(u0.x & 0xffff)) * hv.x
          + bf16_to_f32((unsigned short)(u0.x >> 16))    * hv.y
          + bf16_to_f32((unsigned short)(u0.y & 0xffff)) * hv.z
          + bf16_to_f32((unsigned short)(u0.y >> 16))    * hv.w;
    acc1 += bf16_to_f32((unsigned short)(u1.x & 0xffff)) * hv.x
          + bf16_to_f32((unsigned short)(u1.x >> 16))    * hv.y
          + bf16_to_f32((unsigned short)(u1.y & 0xffff)) * hv.z
          + bf16_to_f32((unsigned short)(u1.y >> 16))    * hv.w;
  }
  __hip_atomic_fetch_add(&m[(size_t)dst * HID + lane],      acc0,
                         __ATOMIC_RELAXED, __HIP_MEMORY_SCOPE_AGENT);
  __hip_atomic_fetch_add(&m[(size_t)dst * HID + lane + 32], acc1,
                         __ATOMIC_RELAXED, __HIP_MEMORY_SCOPE_AGENT);
}

// ---- GRU update: gx = m@W_gih^T, gh = h@W_ghh^T, gate math, h_out ----------
// Wave per 16 nodes. N=192 cols; r/z/n tiles (t, t+4, t+8) align elementwise.
__global__ void gru_kernel(const float* __restrict__ m, const float* __restrict__ h_in,
                           float* __restrict__ h_out,
                           const unsigned short* __restrict__ Wgpk,
                           const unsigned short* __restrict__ Whpk,
                           const float* __restrict__ bgih, const float* __restrict__ bghh) {
  int tile = blockIdx.x, lane = threadIdx.x;
  const float* msrc = m    + (size_t)tile * 16 * HID;
  const float* hsrc = h_in + (size_t)tile * 16 * HID;
  v16bf am0 = make_apack_f32(msrc, HID, 0, lane, HID);
  v16bf am1 = make_apack_f32(msrc, HID, 1, lane, HID);
  v16bf ah0 = make_apack_f32(hsrc, HID, 0, lane, HID);
  v16bf ah1 = make_apack_f32(hsrc, HID, 1, lane, HID);
  int rbase = (lane >= 16) ? 8 : 0;
#pragma unroll
  for (int t = 0; t < 4; ++t) {
    v8f xr = Z8, xz = Z8, xn = Z8, hr = Z8, hz = Z8, hn = Z8;
    xr = wmma_bf16(am0, load_bpack(Wgpk, t,     0, 2, lane), xr);
    xr = wmma_bf16(am1, load_bpack(Wgpk, t,     1, 2, lane), xr);
    xz = wmma_bf16(am0, load_bpack(Wgpk, t + 4, 0, 2, lane), xz);
    xz = wmma_bf16(am1, load_bpack(Wgpk, t + 4, 1, 2, lane), xz);
    xn = wmma_bf16(am0, load_bpack(Wgpk, t + 8, 0, 2, lane), xn);
    xn = wmma_bf16(am1, load_bpack(Wgpk, t + 8, 1, 2, lane), xn);
    hr = wmma_bf16(ah0, load_bpack(Whpk, t,     0, 2, lane), hr);
    hr = wmma_bf16(ah1, load_bpack(Whpk, t,     1, 2, lane), hr);
    hz = wmma_bf16(ah0, load_bpack(Whpk, t + 4, 0, 2, lane), hz);
    hz = wmma_bf16(ah1, load_bpack(Whpk, t + 4, 1, 2, lane), hz);
    hn = wmma_bf16(ah0, load_bpack(Whpk, t + 8, 0, 2, lane), hn);
    hn = wmma_bf16(ah1, load_bpack(Whpk, t + 8, 1, 2, lane), hn);
    int col = t * 16 + (lane & 15);
    float bxr = bgih[col], bxz = bgih[64 + col], bxn = bgih[128 + col];
    float bhr = bghh[col], bhz = bghh[64 + col], bhn = bghh[128 + col];
#pragma unroll
    for (int v = 0; v < 8; ++v) {
      int node = tile * 16 + rbase + v;
      float hold = h_in[(size_t)node * HID + col];
      float r  = sigmoidf_(xr[v] + bxr + hr[v] + bhr);
      float z  = sigmoidf_(xz[v] + bxz + hz[v] + bhz);
      float nn = tanhf(xn[v] + bxn + r * (hn[v] + bhn));
      h_out[(size_t)node * HID + col] = (1.f - z) * nn + z * hold;
    }
  }
}

// ---- Set2Set attention: per-graph softmax + weighted sum (contiguous batch)
__global__ void s2s_attn_kernel(const float* __restrict__ h, const float* __restrict__ hs,
                                float* __restrict__ rs) {
  int g = blockIdx.x, t = threadIdx.x;            // block 256
  int start = (g * N_NODES + N_GRAPHS - 1) / N_GRAPHS;
  int end   = ((g + 1) * N_NODES + N_GRAPHS - 1) / N_GRAPHS;
  int cnt = end - start;
  __shared__ float q[HID];
  __shared__ float ev[160];
  __shared__ float red[256];
  if (t < HID) q[t] = hs[g * HID + t];
  __syncthreads();
  float mymax = -1e30f;
  for (int i = t; i < cnt; i += 256) {
    const float* hr = h + (size_t)(start + i) * HID;
    float s = 0.f;
#pragma unroll 8
    for (int j = 0; j < HID; ++j) s += hr[j] * q[j];
    ev[i] = s;
    mymax = fmaxf(mymax, s);
  }
  red[t] = mymax; __syncthreads();
  for (int o = 128; o > 0; o >>= 1) { if (t < o) red[t] = fmaxf(red[t], red[t + o]); __syncthreads(); }
  float mx = red[0]; __syncthreads();
  float mysum = 0.f;
  for (int i = t; i < cnt; i += 256) { float x = __expf(ev[i] - mx); ev[i] = x; mysum += x; }
  red[t] = mysum; __syncthreads();
  for (int o = 128; o > 0; o >>= 1) { if (t < o) red[t] += red[t + o]; __syncthreads(); }
  float denom = red[0] + 1e-16f;
  if (t < HID) {
    float acc = 0.f;
    for (int i = 0; i < cnt; ++i) acc += ev[i] * h[(size_t)(start + i) * HID + t];
    rs[g * HID + t] = acc / denom;
  }
}

// ---- Set2Set LSTM step (64 graphs x 64 dims, trivial) ----------------------
__global__ void s2s_lstm_kernel(const float* __restrict__ rs, float* __restrict__ hs,
                                float* __restrict__ cs,
                                const float* __restrict__ Wlih, const float* __restrict__ Wlhh,
                                const float* __restrict__ blih, const float* __restrict__ blhh) {
  int g = blockIdx.x, d = threadIdx.x;            // block 64
  __shared__ float x[2 * HID];
  __shared__ float hv[HID];
  float hsd = hs[g * HID + d];
  x[d] = hsd; x[HID + d] = rs[g * HID + d]; hv[d] = hsd;
  __syncthreads();
  float gate[4];
#pragma unroll
  for (int k = 0; k < 4; ++k) {
    int row = k * HID + d;
    float s = blih[row] + blhh[row];
    const float* wi = Wlih + (size_t)row * 2 * HID;
    const float* wh = Wlhh + (size_t)row * HID;
    for (int j = 0; j < 2 * HID; ++j) s += wi[j] * x[j];
    for (int j = 0; j < HID; ++j)     s += wh[j] * hv[j];
    gate[k] = s;
  }
  float ig = sigmoidf_(gate[0]), fg = sigmoidf_(gate[1]);
  float gg = tanhf(gate[2]),     og = sigmoidf_(gate[3]);
  float c = fg * cs[g * HID + d] + ig * gg;
  cs[g * HID + d] = c;
  hs[g * HID + d] = og * tanhf(c);
}

// ---- readout: relu([hs,rs]@W_m1 + b_m1)@W_m2 + b_m2 ------------------------
__global__ void readout_kernel(const float* __restrict__ hs, const float* __restrict__ rs,
                               const float* __restrict__ Wm1, const float* __restrict__ bm1,
                               const float* __restrict__ Wm2, const float* __restrict__ bm2,
                               float* __restrict__ out) {
  int g = blockIdx.x, j = threadIdx.x;            // block 64
  __shared__ float ge[2 * HID];
  __shared__ float hid[HID];
  ge[j] = hs[g * HID + j]; ge[HID + j] = rs[g * HID + j];
  __syncthreads();
  float s = bm1[j];
  for (int k = 0; k < 2 * HID; ++k) s += ge[k] * Wm1[(size_t)k * HID + j];
  hid[j] = fmaxf(s, 0.f);
  __syncthreads();
  if (j == 0) {
    float o = bm2[0];
    for (int k = 0; k < HID; ++k) o += hid[k] * Wm2[k];
    out[g] = o;
  }
}

extern "C" void kernel_launch(void* const* d_in, const int* in_sizes, int n_in,
                              void* d_out, int out_size, void* d_ws, size_t ws_size,
                              hipStream_t stream) {
  const float* node_features = (const float*)d_in[0];
  const float* edge_features = (const float*)d_in[1];
  const int*   edge_index    = (const int*)d_in[2];
  const float* W_proj = (const float*)d_in[4];
  const float* b_proj = (const float*)d_in[5];
  const float* W_e1   = (const float*)d_in[6];
  const float* b_e1   = (const float*)d_in[7];
  const float* W_e2   = (const float*)d_in[8];
  const float* b_e2   = (const float*)d_in[9];
  const float* W_gih  = (const float*)d_in[10];
  const float* W_ghh  = (const float*)d_in[11];
  const float* b_gih  = (const float*)d_in[12];
  const float* b_ghh  = (const float*)d_in[13];
  const float* W_lih  = (const float*)d_in[14];
  const float* W_lhh  = (const float*)d_in[15];
  const float* b_lih  = (const float*)d_in[16];
  const float* b_lhh  = (const float*)d_in[17];
  const float* W_m1   = (const float*)d_in[18];
  const float* b_m1   = (const float*)d_in[19];
  const float* W_m2   = (const float*)d_in[20];
  const float* b_m2   = (const float*)d_in[21];
  float* out = (float*)d_out;

  // Workspace carve (total ~431 MB, dominated by bf16 A)
  char* p = (char*)d_ws;
  auto take = [&](size_t bytes) -> char* {
    char* r = p; p += (bytes + 255) & ~(size_t)255; return r;
  };
  float* h_a = (float*)take((size_t)N_NODES * HID * 4);
  float* h_b = (float*)take((size_t)N_NODES * HID * 4);
  float* m   = (float*)take((size_t)N_NODES * HID * 4);
  unsigned short* hidden  = (unsigned short*)take((size_t)N_EDGES * EDGE_HID * 2);
  unsigned short* Wprojpk = (unsigned short*)take((size_t)4 * 1 * 512 * 2);
  unsigned short* We1pk   = (unsigned short*)take((size_t)8 * 1 * 512 * 2);
  unsigned short* We2pk   = (unsigned short*)take((size_t)256 * 4 * 512 * 2);
  unsigned short* Wgpk    = (unsigned short*)take((size_t)12 * 2 * 512 * 2);
  unsigned short* Whpk    = (unsigned short*)take((size_t)12 * 2 * 512 * 2);
  float* s2s = (float*)take((size_t)3 * N_GRAPHS * HID * 4);
  float* hs = s2s;
  float* cs = s2s + N_GRAPHS * HID;
  float* rs = s2s + 2 * N_GRAPHS * HID;
  unsigned short* A = (unsigned short*)take((size_t)N_EDGES * HID * HID * 2);

  // 1) pre-pack all WMMA B operands (bf16, wave32 layout)
  pack_b_kernel<<<(4 * 512 + 255) / 256, 256, 0, stream>>>(W_proj, Wprojpk, 32, 64, 0, 32);
  pack_b_kernel<<<(8 * 512 + 255) / 256, 256, 0, stream>>>(W_e1, We1pk, 16, 128, 0, 32);
  pack_b_kernel<<<(256 * 4 * 512 + 255) / 256, 256, 0, stream>>>(W_e2, We2pk, 128, 4096, 0, 128);
  pack_b_kernel<<<(24 * 512 + 255) / 256, 256, 0, stream>>>(W_gih, Wgpk, 192, 64, 1, 64);
  pack_b_kernel<<<(24 * 512 + 255) / 256, 256, 0, stream>>>(W_ghh, Whpk, 192, 64, 1, 64);

  // 2) h0 = nf @ W_proj + b   (625 node tiles, one wave each)
  node_proj_kernel<<<N_NODES / 16, 32, 0, stream>>>(node_features, b_proj, Wprojpk, h_a);

  // 3) hidden = relu(ef @ W_e1 + b)  (3125 edge tiles)
  edge_hidden_kernel<<<N_EDGES / 16, 32, 0, stream>>>(edge_features, b_e1, We1pk, hidden);

  // 4) A = hidden @ W_e2 + b -> bf16 (E x 4096); TDM-fed WMMA GEMM
  edge_A_kernel<<<dim3((N_ETILES + 7) / 8, 64), 256, 0, stream>>>(hidden, We2pk, b_e2, A);

  // 5) T message-passing rounds (HBM-bound re-read of bf16 A)
  float* h_cur = h_a;
  float* h_nxt = h_b;
  for (int t = 0; t < T_STEPS; ++t) {
    zero_kernel<<<(N_NODES * HID + 255) / 256, 256, 0, stream>>>(m, N_NODES * HID);
    message_kernel<<<N_EDGES / 8, 256, 0, stream>>>(A, h_cur, edge_index, m);
    gru_kernel<<<N_NODES / 16, 32, 0, stream>>>(m, h_cur, h_nxt, Wgpk, Whpk, b_gih, b_ghh);
    float* tmp = h_cur; h_cur = h_nxt; h_nxt = tmp;
  }

  // 6) Set2Set readout
  zero_kernel<<<(3 * N_GRAPHS * HID + 255) / 256, 256, 0, stream>>>(s2s, 3 * N_GRAPHS * HID);
  for (int t = 0; t < S2S_STEPS; ++t) {
    s2s_attn_kernel<<<N_GRAPHS, 256, 0, stream>>>(h_cur, hs, rs);
    s2s_lstm_kernel<<<N_GRAPHS, 64, 0, stream>>>(rs, hs, cs, W_lih, W_lhh, b_lih, b_lhh);
  }
  readout_kernel<<<N_GRAPHS, 64, 0, stream>>>(hs, rs, W_m1, b_m1, W_m2, b_m2, out);
}
